// KVStore_28647431864944
// MI455X (gfx1250) — matrix operand order repaired and checked by reading, not compiled, not run on hardware
//
#include <hip/hip_runtime.h>
#include <cstdint>
#include <cstddef>

#define KEY_DIM      128
#define VALUE_DIM    128
#define ROW_DIM      256
#define TOP_K        32
#define STORAGE_SIZE 262144
#define BATCH        1024
#define NCHUNK       32
#define CHUNK        (STORAGE_SIZE / NCHUNK)   // 8192 storage rows per chunk
#define BTILE        64                        // batch rows per block (4 waves x 16)
#define STILE        32                        // storage rows per iteration (2 N-subtiles)
#define ITERS        (CHUNK / STILE)           // 256
#define NCAND        (NCHUNK * TOP_K)          // 1024 candidates per batch row
#define EPSSQ        1e-24f                    // rsqrt(max(s,1e-24)) == 1/max(sqrt(s),1e-12)

typedef __attribute__((ext_vector_type(16))) __bf16       v16bf;
typedef __attribute__((ext_vector_type(8)))  float        v8f;
typedef __attribute__((ext_vector_type(4)))  float        f32x4;
typedef __attribute__((ext_vector_type(4)))  unsigned int u32x4;
typedef __attribute__((ext_vector_type(2)))  unsigned int u32x2;

union Frag  { v16bf v; unsigned short us[16]; u32x4 q[2]; };
union Pack8 { __bf16 h[8]; u32x4 q; };
union Pack4 { __bf16 h[4]; u32x2 u; };

// ---------------------------------------------------------------------------
// Kernel 1: L2-normalize queries, emit bf16 row-major. One wave per row.
// ---------------------------------------------------------------------------
__global__ void __launch_bounds__(256)
kv_normalize_q(const float* __restrict__ x, unsigned short* __restrict__ qbf) {
    int gtid = blockIdx.x * blockDim.x + threadIdx.x;
    int row  = gtid >> 5;
    int lane = gtid & 31;
    if (row >= BATCH) return;

    f32x4 v = *reinterpret_cast<const f32x4*>(x + (size_t)row * KEY_DIM + lane * 4);
    float s = v[0]*v[0] + v[1]*v[1] + v[2]*v[2] + v[3]*v[3];
    for (int off = 16; off > 0; off >>= 1) s += __shfl_xor(s, off, 32);
    float inv = rsqrtf(fmaxf(s, EPSSQ));        // v_rsq_f32, no fp32 div chain

    Pack4 p;
    p.h[0] = (__bf16)(v[0] * inv);
    p.h[1] = (__bf16)(v[1] * inv);
    p.h[2] = (__bf16)(v[2] * inv);
    p.h[3] = (__bf16)(v[3] * inv);
    *reinterpret_cast<u32x2*>(qbf + (size_t)row * KEY_DIM + lane * 4) = p.u;
}

// ---------------------------------------------------------------------------
// Kernel 2: fused  sims = Q @ normalize(K)^T  +  per-chunk top-32.
//
// Block = 128 threads (4 waves), covers 64 batch rows x CHUNK storage rows.
// Per iteration (32 storage rows):
//   phase A: 128 threads cooperatively stage the 32x128 fp32 key tile as
//            bf16 in LDS (4 threads per row, 32 elems each) and compute
//            per-row inverse norms (2-step shuffle reduction + v_rsq).
//   phase B: each wave computes TWO 16x16 sims subtiles for its 16 batch
//            rows: 8 v_wmma_f32_16x16x32_bf16 in two independent
//            accumulator chains (A fragments reused across both N tiles),
//            scales by key inverse norms, stages to the sims tile.
//   phase C: 64 owner threads (one per batch row) run threshold top-32
//            maintenance over their 32 new candidates.
// ---------------------------------------------------------------------------
__global__ void __launch_bounds__(128)
kv_sims_topk(const float* __restrict__ storage, const unsigned short* __restrict__ qbf,
             float* __restrict__ cval, int* __restrict__ cidx) {
    __shared__ __align__(16) unsigned short kbuf[STILE * KEY_DIM];  // 8 KB bf16 key tile
    __shared__ float knorm[STILE];
    __shared__ float simtile[BTILE][36];                            // padded vs bank conflicts
    __shared__ float tkv[BTILE][TOP_K];
    __shared__ int   tki[BTILE][TOP_K];

    const int tid  = threadIdx.x;
    const int lane = tid & 31;
    const int wave = tid >> 5;             // 0..3
    const int half = (lane >> 4) & 1;
    const int l16  = lane & 15;
    const int batchbase = blockIdx.x * BTILE;
    const int chunkbase = blockIdx.y * CHUNK;

    for (int e = tid; e < BTILE * TOP_K; e += 128) {
        tkv[e >> 5][e & 31] = -1e30f;
        tki[e >> 5][e & 31] = 0;
    }
    float rowmin = -1e30f;                 // valid only for owner threads tid<64

    // A fragments: this wave's 16 query rows, all of K=0..127, loaded once.
    Frag afrag[4];
    {
        const unsigned short* qrow =
            qbf + (size_t)(batchbase + wave * 16 + l16) * KEY_DIM;
        const int koff = half * 8;
#pragma unroll
        for (int kb = 0; kb < 4; ++kb) {
            afrag[kb].q[0] = *reinterpret_cast<const u32x4*>(qrow + kb * 32 + koff);
            afrag[kb].q[1] = *reinterpret_cast<const u32x4*>(qrow + kb * 32 + 16 + koff);
        }
    }

    const int r32 = tid >> 2;              // staging: key row 0..31
    const int seg = tid & 3;               // staging: 32-element segment 0..3

    __syncthreads();

    for (int it = 0; it < ITERS; ++it) {
        const int tilebase = chunkbase + it * STILE;

        // ---- phase A: stage key tile (fp32 -> bf16) + inverse norms ----
        {
            const float* src = storage + (size_t)(tilebase + r32) * ROW_DIM + seg * 32;
            if (it + 1 < ITERS)
                __builtin_prefetch(src + (size_t)STILE * ROW_DIM, 0, 0);  // global_prefetch_b8

            float sum = 0.0f;
            unsigned short* dst = kbuf + r32 * KEY_DIM + seg * 32;
#pragma unroll
            for (int j = 0; j < 4; ++j) {
                f32x4 fa = *reinterpret_cast<const f32x4*>(src + j * 8);
                f32x4 fb = *reinterpret_cast<const f32x4*>(src + j * 8 + 4);
                sum += fa[0]*fa[0] + fa[1]*fa[1] + fa[2]*fa[2] + fa[3]*fa[3]
                     + fb[0]*fb[0] + fb[1]*fb[1] + fb[2]*fb[2] + fb[3]*fb[3];
                Pack8 p;
#pragma unroll
                for (int i = 0; i < 4; ++i) {
                    p.h[i]     = (__bf16)fa[i];
                    p.h[i + 4] = (__bf16)fb[i];
                }
                *reinterpret_cast<u32x4*>(dst + j * 8) = p.q;
            }

            // 4 threads share a key row: butterfly over seg bits
            sum += __shfl_xor(sum, 2, 32);
            sum += __shfl_xor(sum, 1, 32);
            if (seg == 0) knorm[r32] = rsqrtf(fmaxf(sum, EPSSQ));
        }
        __syncthreads();

        // ---- phase B: 8 WMMAs per wave (2 N-subtiles x 4 K-steps) ----
        {
            v8f c0, c1;
#pragma unroll
            for (int r = 0; r < 8; ++r) { c0[r] = 0.0f; c1[r] = 0.0f; }

#pragma unroll
            for (int kb = 0; kb < 4; ++kb) {
                const int ko = kb * 32 + half * 8;
                Frag b0, b1;
                const unsigned short* bp0 = kbuf + l16 * KEY_DIM + ko;
                const unsigned short* bp1 = kbuf + (16 + l16) * KEY_DIM + ko;
                b0.q[0] = *reinterpret_cast<const u32x4*>(bp0);
                b0.q[1] = *reinterpret_cast<const u32x4*>(bp0 + 16);
                b1.q[0] = *reinterpret_cast<const u32x4*>(bp1);
                b1.q[1] = *reinterpret_cast<const u32x4*>(bp1 + 16);
                c0 = __builtin_amdgcn_wmma_f32_16x16x32_bf16(false, afrag[kb].v,
                                                             false, b0.v,
                                                             (short)0, c0, false, false);
                c1 = __builtin_amdgcn_wmma_f32_16x16x32_bf16(false, afrag[kb].v,
                                                             false, b1.v,
                                                             (short)0, c1, false, false);
            }

            const float invk0 = knorm[l16];
            const float invk1 = knorm[16 + l16];
            const int   m     = wave * 16 + half * 8;
#pragma unroll
            for (int r = 0; r < 8; ++r) {
                simtile[m + r][l16]      = c0[r] * invk0;
                simtile[m + r][16 + l16] = c1[r] * invk1;
            }
        }
        __syncthreads();

        // ---- phase C: threshold top-32 maintenance, one owner per row ----
        if (tid < BTILE) {
            const int m = tid;
#pragma unroll 4
            for (int col = 0; col < STILE; ++col) {
                const float v = simtile[m][col];
                if (v > rowmin) {
                    int pos = 0; float mn = tkv[m][0];
                    for (int k = 1; k < TOP_K; ++k) {
                        float tv = tkv[m][k];
                        if (tv < mn) { mn = tv; pos = k; }
                    }
                    tkv[m][pos] = v; tki[m][pos] = tilebase + col;
                    mn = tkv[m][0];
                    for (int k = 1; k < TOP_K; ++k) mn = fminf(mn, tkv[m][k]);
                    rowmin = mn;
                }
            }
        }
        __syncthreads();
    }

    // emit this chunk's candidates (64 rows x 32 entries)
    for (int e = tid; e < BTILE * TOP_K; e += 128) {
        const int m = e >> 5, k = e & 31;
        const size_t o = ((size_t)(batchbase + m) * NCHUNK + blockIdx.y) * TOP_K + k;
        cval[o] = tkv[m][k];
        cidx[o] = tki[m][k];
    }
}

// ---------------------------------------------------------------------------
// Kernel 3: per batch row — merge 1024 candidates -> exact top-32 by rank
// selection (indices globally unique => strict total order), softmax,
// weighted coalesced gather of the 128-wide value rows.
// ---------------------------------------------------------------------------
__global__ void __launch_bounds__(128)
kv_final(const float* __restrict__ storage, const float* __restrict__ cval,
         const int* __restrict__ cidx, float* __restrict__ out) {
    __shared__ float sv[NCAND];
    __shared__ int   si[NCAND];
    __shared__ float wv[TOP_K];
    __shared__ int   wi[TOP_K];
    __shared__ float wexp[TOP_K];
    __shared__ float ssum;

    const int b   = blockIdx.x;
    const int tid = threadIdx.x;

    for (int i = tid; i < NCAND; i += 128) {
        const size_t o = (size_t)b * NCAND + i;
        sv[i] = cval[o];
        si[i] = cidx[o];
    }
    __syncthreads();

#pragma unroll
    for (int t = 0; t < NCAND / 128; ++t) {
        const int   i  = tid + t * 128;
        const float v  = sv[i];
        const int   id = si[i];
        int rank = 0;
        for (int j = 0; j < NCAND; ++j) {
            const float vj = sv[j];
            rank += (vj > v) || (vj == v && si[j] < id);
        }
        if (rank < TOP_K) { wv[rank] = v; wi[rank] = id; }
    }
    __syncthreads();

    if (tid < TOP_K) wexp[tid] = expf(wv[tid] - wv[0]);  // wv[0] is the max
    __syncthreads();
    if (tid == 0) {
        float s = 0.0f;
        for (int k = 0; k < TOP_K; ++k) s += wexp[k];
        ssum = s;
    }
    __syncthreads();

    const int d = tid;                     // blockDim == VALUE_DIM
    float acc = 0.0f;
    for (int k = 0; k < TOP_K; ++k)
        acc += wexp[k] * storage[(size_t)wi[k] * ROW_DIM + KEY_DIM + d];
    out[(size_t)b * VALUE_DIM + d] = acc / ssum;
}

// ---------------------------------------------------------------------------
extern "C" void kernel_launch(void* const* d_in, const int* in_sizes, int n_in,
                              void* d_out, int out_size, void* d_ws, size_t ws_size,
                              hipStream_t stream) {
    const float* x       = (const float*)d_in[0];
    const float* storage = (const float*)d_in[1];
    float*       out     = (float*)d_out;

    char* p = (char*)d_ws;
    unsigned short* qbf = (unsigned short*)p;                                    // 256 KB
    float* cval = (float*)(p + (size_t)BATCH * KEY_DIM * sizeof(unsigned short));// 4 MB
    int*   cidx = (int*)  (p + (size_t)BATCH * KEY_DIM * sizeof(unsigned short)
                             + (size_t)BATCH * NCAND * sizeof(float));           // 4 MB

    kv_normalize_q<<<(BATCH * 32) / 256, 256, 0, stream>>>(x, qbf);

    dim3 grid(BATCH / BTILE, NCHUNK);
    kv_sims_topk<<<grid, 128, 0, stream>>>(storage, qbf, cval, cidx);

    kv_final<<<BATCH, 128, 0, stream>>>(storage, cval, cidx, out);
}
